// PointToDense_16664473108566
// MI455X (gfx1250) — compile-verified, matrix operand order checked by reading, not run on hardware
//
#include <hip/hip_runtime.h>
#include <hip/hip_bf16.h>
#include <math.h>

// ---------------------------------------------------------------------------
// PointToDense (DGCNN encoder + TimeMerge) for MI455X / gfx1250.
// Heavy 1x1-conv / linear layers -> v_wmma_f32_16x16x32_f16.
// Weights pre-packed to zero-padded f16 so the A tile is staged into LDS with
// global_load_async_to_lds_b128 (CDNA5 async DMA, ASYNCcnt). BatchNorm layers
// use a two-pass GEMM (stats pass, then fused BN/GELU pass).
// ---------------------------------------------------------------------------

#define B_  4
#define N_  4096
#define K_  10

typedef __attribute__((ext_vector_type(16))) _Float16 v16h;
typedef __attribute__((ext_vector_type(8)))  _Float16 h8;
typedef __attribute__((ext_vector_type(8)))  float    v8f;

__device__ __forceinline__ float gelu_f(float x) {
    return 0.5f * x * (1.0f + erff(x * 0.70710678118654752f));
}
__device__ __forceinline__ float silu_f(float x) {
    return x * (1.0f / (1.0f + __expf(-x)));
}
// Ordered-int trick for float atomic max (works for mixed signs).
__device__ __forceinline__ void atomicMaxFloat(float* addr, float val) {
    if (val >= 0.0f) atomicMax((int*)addr, __float_as_int(val));
    else             atomicMin((unsigned int*)addr, __float_as_uint(val));
}

// ---------------------------------------------------------------------------
__global__ void init_kernel(float* __restrict__ xcat, float* __restrict__ stats,
                            long nx, int ns) {
    long t = (long)blockIdx.x * 256 + threadIdx.x;
    if (t < nx) xcat[t] = -1.0e30f;
    if (t < ns) stats[t] = 0.0f;
}

// Zero-padded f32 -> f16 weight pack: Wh[Mp,Kp] from W[M,Kd].
__global__ void packw_kernel(const float* __restrict__ W, _Float16* __restrict__ Wh,
                             int M, int Kd, int Mp, int Kp) {
    int t = blockIdx.x * 256 + threadIdx.x;
    if (t >= Mp * Kp) return;
    int m = t / Kp, k = t % Kp;
    Wh[t] = (m < M && k < Kd) ? (_Float16)W[(long)m * Kd + k] : (_Float16)0.0f;
}

// xx[b][n] = sum_c x[b][c][n]^2
__global__ void rowsumsq_kernel(const float* __restrict__ x, float* __restrict__ xx,
                                int total, int N, int C, int bstride, int cstride) {
    int t = blockIdx.x * 256 + threadIdx.x;
    if (t >= total) return;
    int b = t / N, n = t % N;
    const float* xb = x + (long)b * bstride;
    float s = 0.0f;
    for (int c = 0; c < C; ++c) { float v = xb[(long)c * cstride + n]; s += v * v; }
    xx[t] = s;
}

// kNN top-10 by neg squared distance; one wave per query point.
template <int C>
__global__ __launch_bounds__(32)
void knn_kernel(const float* __restrict__ x, const float* __restrict__ xx,
                int* __restrict__ idx, int N, int bstride, int cstride) {
    int b = blockIdx.y, n = blockIdx.x, lane = threadIdx.x;
    const float* xb = x + (long)b * bstride;
    float q[C];
#pragma unroll
    for (int c = 0; c < C; ++c) q[c] = xb[(long)c * cstride + n];
    float xxq = xx[b * N + n];
    float bv[10]; int bi[10];
#pragma unroll
    for (int t = 0; t < 10; ++t) { bv[t] = -3.0e38f; bi[t] = 0; }
    for (int m = lane; m < N; m += 32) {
        float dot = 0.0f;
#pragma unroll
        for (int c = 0; c < C; ++c) dot += q[c] * xb[(long)c * cstride + m];
        float nd = 2.0f * dot - xxq - xx[b * N + m];
        if (nd > bv[9]) {
            int p = 9;
            while (p > 0 && nd > bv[p - 1]) { bv[p] = bv[p - 1]; bi[p] = bi[p - 1]; --p; }
            bv[p] = nd; bi[p] = m;
        }
    }
    __shared__ float sv[320];
    __shared__ int   si[320];
    for (int t = 0; t < 10; ++t) { sv[lane * 10 + t] = bv[t]; si[lane * 10 + t] = bi[t]; }
    __syncthreads();
    if (lane == 0) {
        for (int t = 0; t < 10; ++t) {
            float best = -3.5e38f; int bj = 0;
            for (int j = 0; j < 320; ++j) {
                if (sv[j] > best || (sv[j] == best && si[j] < si[bj])) { best = sv[j]; bj = j; }
            }
            idx[((long)b * N + n) * 10 + t] = si[bj];
            sv[bj] = -3.5e38f;
        }
    }
}

// conv1 layer A: on-the-fly edge features (4ch) -> 11ch. PASS0 stats, PASS1 BN+GELU.
template <int PASS>
__global__ __launch_bounds__(256)
void conv1a_kernel(const float* __restrict__ xy, const int* __restrict__ idx1,
                   const float* __restrict__ w, const float* __restrict__ bias,
                   const float* __restrict__ g, const float* __restrict__ beta,
                   float* __restrict__ sum, float* __restrict__ sq,
                   float* __restrict__ out, float invCnt) {
    __shared__ float sS[16], sQ[16];
    int tid = threadIdx.x;
    if (PASS == 0 && tid < 16) { sS[tid] = 0.0f; sQ[tid] = 0.0f; }
    __syncthreads();
    const int SNK = N_ * K_;
    int t = blockIdx.x * 256 + tid;
    if (t < B_ * SNK) {
        int b = t / SNK, r = t % SNK, n = r / K_;
        int j = idx1[t];
        const float* xb = xy + (long)b * 2 * N_;
        float f0 = xb[j] - xb[n], f1 = xb[N_ + j] - xb[N_ + n];
        float f2 = xb[n], f3 = xb[N_ + n];
        for (int o = 0; o < 11; ++o) {
            float y = w[o * 4 + 0] * f0 + w[o * 4 + 1] * f1 +
                      w[o * 4 + 2] * f2 + w[o * 4 + 3] * f3 + bias[o];
            if (PASS == 0) { atomicAdd(&sS[o], y); atomicAdd(&sQ[o], y * y); }
            else {
                float mean = sum[o] * invCnt;
                float var  = sq[o] * invCnt - mean * mean;
                float act  = gelu_f((y - mean) * g[o] * rsqrtf(var + 1e-5f) + beta[o]);
                out[((long)b * 11 + o) * SNK + r] = act;
            }
        }
    }
    if (PASS == 0) {
        __syncthreads();
        if (tid < 11) { atomicAdd(&sum[tid], sS[tid]); atomicAdd(&sq[tid], sQ[tid]); }
    }
}

// conv1 layer B: 11 -> 32, max over k via atomicMax into xcat rows 0..31.
template <int PASS>
__global__ __launch_bounds__(256)
void conv1b_kernel(const float* __restrict__ h1,
                   const float* __restrict__ w, const float* __restrict__ bias,
                   const float* __restrict__ g, const float* __restrict__ beta,
                   float* __restrict__ sum, float* __restrict__ sq,
                   float* __restrict__ xcat, float invCnt) {
    __shared__ float sS[32], sQ[32];
    int tid = threadIdx.x;
    if (PASS == 0 && tid < 32) { sS[tid] = 0.0f; sQ[tid] = 0.0f; }
    __syncthreads();
    const int SNK = N_ * K_;
    int t = blockIdx.x * 256 + tid;
    if (t < B_ * SNK) {
        int b = t / SNK, r = t % SNK, n = r / K_;
        float h[11];
#pragma unroll
        for (int c = 0; c < 11; ++c) h[c] = h1[((long)b * 11 + c) * SNK + r];
        for (int o = 0; o < 32; ++o) {
            float y = bias[o];
#pragma unroll
            for (int c = 0; c < 11; ++c) y += w[o * 11 + c] * h[c];
            if (PASS == 0) { atomicAdd(&sS[o], y); atomicAdd(&sQ[o], y * y); }
            else {
                float mean = sum[o] * invCnt;
                float var  = sq[o] * invCnt - mean * mean;
                float act  = gelu_f((y - mean) * g[o] * rsqrtf(var + 1e-5f) + beta[o]);
                atomicMaxFloat(&xcat[((long)b * 544 + o) * N_ + n], act);
            }
        }
    }
    if (PASS == 0) {
        __syncthreads();
        if (tid < 32) { atomicAdd(&sum[tid], sS[tid]); atomicAdd(&sq[tid], sQ[tid]); }
    }
}

// Edge features for conv2: x1 (xcat rows 0..31) -> ef2 [B,64,N*K]
__global__ __launch_bounds__(256)
void edge2_kernel(const float* __restrict__ xcat, const int* __restrict__ idx2,
                  float* __restrict__ ef2) {
    const int SNK = N_ * K_;
    int t = blockIdx.x * 256 + threadIdx.x;
    if (t >= B_ * SNK) return;
    int b = t / SNK, r = t % SNK, n = r / K_;
    int j = idx2[t];
    for (int c = 0; c < 32; ++c) {
        float xc = xcat[((long)b * 544 + c) * N_ + n];
        float xj = xcat[((long)b * 544 + c) * N_ + j];
        ef2[((long)b * 64 + c) * SNK + r]      = xj - xc;
        ef2[((long)b * 64 + 32 + c) * SNK + r] = xc;
    }
}

// ---------------------------------------------------------------------------
// WMMA GEMM:  Y[M,S] = W[M,Kd] @ X[Kd,S]  per batch (blockIdx.z).
// 256 threads = 8 waves; block tile 64(M) x 64(S); K chunk 32.
// A tile: pre-packed f16 weights copied global->LDS via async DMA
//         (global_load_async_to_lds_b128, s_wait_asynccnt).
// B tile: f32 activations, float4 loads, cvt to f16, stored transposed [N][K]
//         so fragments are two contiguous ds_load_b128 per operand.
// MODE 0: BN stats.  MODE 1: BN+GELU store f32.  MODE 2: BN+GELU atomicMax
// over k-groups.  MODE 3: plain Wx+b store f32.
// ---------------------------------------------------------------------------
template <int MODE>
__global__ __launch_bounds__(256)
void gemm_wmma(const _Float16* __restrict__ Wh, const float* __restrict__ X,
               const float* __restrict__ bias, int M, int Kd, int Kp, int S,
               float* __restrict__ outF32,
               float* __restrict__ statSum, float* __restrict__ statSq,
               const float* __restrict__ g, const float* __restrict__ beta,
               float invCnt,
               float* __restrict__ maxBase, int maxBstride, int maxChanOff,
               int maxN, int kGroup) {
    __shared__ _Float16 As[64][40];    // [M][K] tile, rows 16B-aligned (80B stride)
    __shared__ _Float16 Bst[64][40];   // [N][K] tile (transposed store)
    __shared__ float sS[64], sQ[64];

    int tid = threadIdx.x;
    int b   = blockIdx.z;
    int m0  = blockIdx.y * 64;
    int s0  = blockIdx.x * 64;
    const float* Xb = X + (long)b * Kd * S;

    if (MODE == 0 && tid < 64) { sS[tid] = 0.0f; sQ[tid] = 0.0f; }

    int wv = tid >> 5, lane = tid & 31;
    int tr = wv >> 2, tc = wv & 3;          // waves: 2 (M) x 4 (S) of 16x16 pairs
    int mA = lane & 15, half = lane >> 4;

    // staging coordinates
    int ra = tid >> 2, ca = (tid & 3) * 8;  // A: 64 rows x 32 halves, 16B/thread
    int rb = tid >> 3, cb = (tid & 7) * 8;  // B: 32 rows x 64 cols, 8 f32/thread

    v8f acc[2] = {};
    for (int kc = 0; kc < Kp; kc += 32) {
        // ---- A tile: pure f16 copy, async DMA straight into LDS ----
        {
            const _Float16* gsrc = Wh + (long)(m0 + ra) * Kp + kc + ca;
            unsigned ldst = (unsigned)(size_t)&As[ra][ca];   // low 32 bits = LDS offset
            unsigned long long ga = (unsigned long long)(size_t)gsrc;
            asm volatile("global_load_async_to_lds_b128 %0, %1, off"
                         :: "v"(ldst), "v"(ga) : "memory");
        }
        // ---- B tile: f32 -> f16, transposed into LDS ----
        {
            int kx = kc + rb;
            if (kx < Kd) {
                const float* src = &Xb[(long)kx * S + s0 + cb];
                float4 p0 = *(const float4*)(src);
                float4 p1 = *(const float4*)(src + 4);
                Bst[cb + 0][rb] = (_Float16)p0.x; Bst[cb + 1][rb] = (_Float16)p0.y;
                Bst[cb + 2][rb] = (_Float16)p0.z; Bst[cb + 3][rb] = (_Float16)p0.w;
                Bst[cb + 4][rb] = (_Float16)p1.x; Bst[cb + 5][rb] = (_Float16)p1.y;
                Bst[cb + 6][rb] = (_Float16)p1.z; Bst[cb + 7][rb] = (_Float16)p1.w;
            } else {
#pragma unroll
                for (int i = 0; i < 8; ++i) Bst[cb + i][rb] = (_Float16)0.0f;
            }
            if (kc + 32 < Kd)  // pull next chunk toward cache
                __builtin_prefetch(&Xb[(long)(kx + 32) * S + s0 + cb], 0, 1);
        }
        asm volatile("s_wait_asynccnt 0" ::: "memory");
        __syncthreads();

        // ---- fragments: two contiguous 16B LDS reads per operand ----
        union F { v16h v; h8 q[2]; } a0, a1, bu;
        int rowA = tr * 16 + mA;
        a0.q[0] = *(const h8*)&As[rowA][half * 8];
        a0.q[1] = *(const h8*)&As[rowA][16 + half * 8];
        a1.q[0] = *(const h8*)&As[rowA + 32][half * 8];
        a1.q[1] = *(const h8*)&As[rowA + 32][16 + half * 8];
        int colN = tc * 16 + mA;
        bu.q[0] = *(const h8*)&Bst[colN][half * 8];
        bu.q[1] = *(const h8*)&Bst[colN][16 + half * 8];

        acc[0] = __builtin_amdgcn_wmma_f32_16x16x32_f16(
            false, a0.v, false, bu.v, (short)0, acc[0], false, false);
        acc[1] = __builtin_amdgcn_wmma_f32_16x16x32_f16(
            false, a1.v, false, bu.v, (short)0, acc[1], false, false);
        __syncthreads();
    }

    int nl = lane & 15;
#pragma unroll
    for (int ha = 0; ha < 2; ++ha) {
#pragma unroll
        for (int r8 = 0; r8 < 8; ++r8) {
            int m_l = tr * 16 + r8 + 8 * half + ha * 32;  // C/D layout: VGPR r -> M=r+8*half
            int m_g = m0 + m_l;
            int s_g = s0 + tc * 16 + nl;
            if (m_g < M) {
                float val = acc[ha][r8] + bias[m_g];
                if (MODE == 0) {
                    atomicAdd(&sS[m_l], val);
                    atomicAdd(&sQ[m_l], val * val);
                } else if (MODE == 3) {
                    outF32[(long)b * M * S + (long)m_g * S + s_g] = val;
                } else {
                    float mean = statSum[m_g] * invCnt;
                    float var  = statSq[m_g] * invCnt - mean * mean;
                    float sc   = g[m_g] * rsqrtf(var + 1e-5f);
                    float act  = gelu_f((val - mean) * sc + beta[m_g]);
                    if (MODE == 1)
                        outF32[(long)b * M * S + (long)m_g * S + s_g] = act;
                    else
                        atomicMaxFloat(maxBase + (long)b * maxBstride +
                                       (long)(maxChanOff + m_g) * maxN + s_g / kGroup, act);
                }
            }
        }
    }
    if (MODE == 0) {
        __syncthreads();
        if (tid < 64 && m0 + tid < M) {
            atomicAdd(&statSum[m0 + tid], sS[tid]);
            atomicAdd(&statSq[m0 + tid], sQ[tid]);
        }
    }
}

// TimeMerge linear: t = gelu(time_emb) @ lin_w^T + lin_b  [B,1024]
__global__ __launch_bounds__(256)
void tlin_kernel(const float* __restrict__ te, const float* __restrict__ w,
                 const float* __restrict__ bias, float* __restrict__ out) {
    int t = blockIdx.x * 256 + threadIdx.x;
    if (t >= B_ * 1024) return;
    int b = t >> 10, o = t & 1023;
    const float* teb = te + b * 256;
    float s = 0.0f;
    for (int i = 0; i < 256; ++i) s += gelu_f(teb[i]) * w[o * 256 + i];
    out[t] = s + bias[o];
}

// RMSNorm per (b,n) column. WHICH=1: modulate+silu -> hact. WHICH=2: silu + dense.
template <int WHICH>
__global__ __launch_bounds__(256)
void rms_kernel(const float* __restrict__ in, const float* __restrict__ g,
                const float* __restrict__ tlin, const float* __restrict__ dense,
                float* __restrict__ out) {
    int col = blockIdx.x;
    int b = col / N_, n = col % N_;
    int c = threadIdx.x;
    float v0 = in[((long)b * 512 + c) * N_ + n];
    float v1 = in[((long)b * 512 + c + 256) * N_ + n];
    __shared__ float red[256];
    red[c] = v0 * v0 + v1 * v1;
    __syncthreads();
    for (int st = 128; st > 0; st >>= 1) {
        if (c < st) red[c] += red[c + st];
        __syncthreads();
    }
    float inv = 22.62741699796952f / fmaxf(sqrtf(red[0]), 1e-12f); // sqrt(512)/||x||
#pragma unroll
    for (int q = 0; q < 2; ++q) {
        int cc = c + q * 256;
        float v = q ? v1 : v0;
        float xn = v * inv * g[cc];
        float r;
        if (WHICH == 1) {
            float scale = tlin[b * 1024 + cc];
            float shift = tlin[b * 1024 + 512 + cc];
            r = silu_f(xn * (scale + 1.0f) + shift);
        } else {
            r = silu_f(xn) + dense[((long)b * 512 + cc) * N_ + n];
        }
        out[((long)b * 512 + cc) * N_ + n] = r;
    }
}

// ---------------------------------------------------------------------------
extern "C" void kernel_launch(void* const* d_in, const int* in_sizes, int n_in,
                              void* d_out, int out_size, void* d_ws, size_t ws_size,
                              hipStream_t stream) {
    (void)in_sizes; (void)n_in; (void)out_size; (void)ws_size;
    const float* xy  = (const float*)d_in[0];
    const float* te  = (const float*)d_in[1];
    const float *w1a = (const float*)d_in[2],  *b1a = (const float*)d_in[3];
    const float *g1a = (const float*)d_in[4],  *be1a = (const float*)d_in[5];
    const float *w1b = (const float*)d_in[6],  *b1b = (const float*)d_in[7];
    const float *g1b = (const float*)d_in[8],  *be1b = (const float*)d_in[9];
    const float *w2a = (const float*)d_in[10], *b2a = (const float*)d_in[11];
    const float *g2a = (const float*)d_in[12], *be2a = (const float*)d_in[13];
    const float *w2b = (const float*)d_in[14], *b2b = (const float*)d_in[15];
    const float *g2bp = (const float*)d_in[16], *be2b = (const float*)d_in[17];
    const float *w3a = (const float*)d_in[18], *b3a = (const float*)d_in[19];
    const float *g3a = (const float*)d_in[20], *be3a = (const float*)d_in[21];
    const float *w3b = (const float*)d_in[22], *b3b = (const float*)d_in[23];
    const float *g3b = (const float*)d_in[24], *be3b = (const float*)d_in[25];
    const float *linw = (const float*)d_in[26], *linb = (const float*)d_in[27];
    const float *tb1w = (const float*)d_in[28], *tb1b = (const float*)d_in[29];
    const float *tb1g = (const float*)d_in[30];
    const float *tb2w = (const float*)d_in[31], *tb2b = (const float*)d_in[32];
    const float *tb2g = (const float*)d_in[33];

    float* ws = (float*)d_ws;
    size_t off = 0;
    auto alloc  = [&](size_t n) { float* p = ws + off; off += n; return p; };
    auto allocH = [&](size_t nh) {
        off = (off + 3) & ~((size_t)3);           // 16B align for b128 loads
        _Float16* p = (_Float16*)(ws + off);
        off += (nh + 1) / 2;
        return p;
    };
    const int SNK = N_ * K_;
    float* xx1   = alloc((size_t)B_ * N_);
    int*   idx1  = (int*)alloc((size_t)B_ * SNK);
    float* a_h1  = alloc((size_t)B_ * 11 * SNK);
    float* xcat  = alloc((size_t)B_ * 544 * N_);
    float* xx2   = alloc((size_t)B_ * N_);
    int*   idx2  = (int*)alloc((size_t)B_ * SNK);
    float* ef2   = alloc((size_t)B_ * 64 * SNK);
    float* a_h2  = alloc((size_t)B_ * 181 * SNK);
    float* a_h3  = alloc((size_t)B_ * 527 * N_);
    float* dense = alloc((size_t)B_ * 512 * N_);
    float* hb    = alloc((size_t)B_ * 512 * N_);
    float* hact  = alloc((size_t)B_ * 512 * N_);
    float* tlin  = alloc((size_t)B_ * 1024);
    float* stats = alloc(12 * 1024);
    // padded f16 weights (Mp = ceil64, Kp = ceil32)
    _Float16* w2ah = allocH(192 * 64);
    _Float16* w2bh = allocH(512 * 192);
    _Float16* w3ah = allocH(576 * 544);
    _Float16* w3bh = allocH(512 * 544);
    _Float16* tb1h = allocH(512 * 512);
    _Float16* tb2h = allocH(512 * 512);

    float *s1a = stats,             *q1a = stats + 1024;
    float *s1b = stats + 2 * 1024,  *q1b = stats + 3 * 1024;
    float *s2a = stats + 4 * 1024,  *q2a = stats + 5 * 1024;
    float *s2b = stats + 6 * 1024,  *q2b = stats + 7 * 1024;
    float *s3a = stats + 8 * 1024,  *q3a = stats + 9 * 1024;
    float *s3b = stats + 10 * 1024, *q3b = stats + 11 * 1024;

    const float invNK = 1.0f / (float)(B_ * SNK);
    const float invBN = 1.0f / (float)(B_ * N_);
    const long  nxcat = (long)B_ * 544 * N_;

    init_kernel<<<(unsigned)((nxcat + 255) / 256), 256, 0, stream>>>(xcat, stats, nxcat, 12 * 1024);

    // pack weights to padded f16
    auto packL = [&](const float* W, _Float16* Wh, int M, int Kd, int Mp, int Kp) {
        int tot = Mp * Kp;
        packw_kernel<<<(tot + 255) / 256, 256, 0, stream>>>(W, Wh, M, Kd, Mp, Kp);
    };
    packL(w2a, w2ah, 181, 64, 192, 64);
    packL(w2b, w2bh, 512, 181, 512, 192);
    packL(w3a, w3ah, 527, 544, 576, 544);
    packL(w3b, w3bh, 512, 527, 512, 544);
    packL(tb1w, tb1h, 512, 512, 512, 512);
    packL(tb2w, tb2h, 512, 512, 512, 512);

    // stage 1: kNN(xy), conv1 4->11->32, max_k -> xcat rows 0..31
    rowsumsq_kernel<<<(B_ * N_ + 255) / 256, 256, 0, stream>>>(xy, xx1, B_ * N_, N_, 2, 2 * N_, N_);
    knn_kernel<2><<<dim3(N_, B_), 32, 0, stream>>>(xy, xx1, idx1, N_, 2 * N_, N_);
    conv1a_kernel<0><<<(B_ * SNK + 255) / 256, 256, 0, stream>>>(xy, idx1, w1a, b1a, g1a, be1a, s1a, q1a, a_h1, invNK);
    conv1a_kernel<1><<<(B_ * SNK + 255) / 256, 256, 0, stream>>>(xy, idx1, w1a, b1a, g1a, be1a, s1a, q1a, a_h1, invNK);
    conv1b_kernel<0><<<(B_ * SNK + 255) / 256, 256, 0, stream>>>(a_h1, w1b, b1b, g1b, be1b, s1b, q1b, xcat, invNK);
    conv1b_kernel<1><<<(B_ * SNK + 255) / 256, 256, 0, stream>>>(a_h1, w1b, b1b, g1b, be1b, s1b, q1b, xcat, invNK);

    // stage 2: kNN(x1), edge features, conv2 64->181->512 (WMMA), max_k -> xcat rows 32..543
    rowsumsq_kernel<<<(B_ * N_ + 255) / 256, 256, 0, stream>>>(xcat, xx2, B_ * N_, N_, 32, 544 * N_, N_);
    knn_kernel<32><<<dim3(N_, B_), 32, 0, stream>>>(xcat, xx2, idx2, N_, 544 * N_, N_);
    edge2_kernel<<<(B_ * SNK + 255) / 256, 256, 0, stream>>>(xcat, idx2, ef2);

    dim3 ga(SNK / 64, 192 / 64, B_);
    gemm_wmma<0><<<ga, 256, 0, stream>>>(w2ah, ef2, b2a, 181, 64, 64, SNK, nullptr, s2a, q2a, nullptr, nullptr, invNK, nullptr, 0, 0, 1, 1);
    gemm_wmma<1><<<ga, 256, 0, stream>>>(w2ah, ef2, b2a, 181, 64, 64, SNK, a_h2, s2a, q2a, g2a, be2a, invNK, nullptr, 0, 0, 1, 1);
    dim3 gb(SNK / 64, 512 / 64, B_);
    gemm_wmma<0><<<gb, 256, 0, stream>>>(w2bh, a_h2, b2b, 512, 181, 192, SNK, nullptr, s2b, q2b, nullptr, nullptr, invNK, nullptr, 0, 0, 1, 1);
    gemm_wmma<2><<<gb, 256, 0, stream>>>(w2bh, a_h2, b2b, 512, 181, 192, SNK, nullptr, s2b, q2b, g2bp, be2b, invNK, xcat, 544 * N_, 32, N_, K_);

    // stage 3: conv3 544->527->512 (1d, WMMA) -> dense
    dim3 gc(N_ / 64, 576 / 64, B_);
    gemm_wmma<0><<<gc, 256, 0, stream>>>(w3ah, xcat, b3a, 527, 544, 544, N_, nullptr, s3a, q3a, nullptr, nullptr, invBN, nullptr, 0, 0, 1, 1);
    gemm_wmma<1><<<gc, 256, 0, stream>>>(w3ah, xcat, b3a, 527, 544, 544, N_, a_h3, s3a, q3a, g3a, be3a, invBN, nullptr, 0, 0, 1, 1);
    dim3 gd(N_ / 64, 512 / 64, B_);
    gemm_wmma<0><<<gd, 256, 0, stream>>>(w3bh, a_h3, b3b, 512, 527, 544, N_, nullptr, s3b, q3b, nullptr, nullptr, invBN, nullptr, 0, 0, 1, 1);
    gemm_wmma<1><<<gd, 256, 0, stream>>>(w3bh, a_h3, b3b, 512, 527, 544, N_, dense, s3b, q3b, g3b, be3b, invBN, nullptr, 0, 0, 1, 1);

    // stage 4: TimeMerge
    tlin_kernel<<<(B_ * 1024 + 255) / 256, 256, 0, stream>>>(te, linw, linb, tlin);
    gemm_wmma<3><<<gd, 256, 0, stream>>>(tb1h, dense, tb1b, 512, 512, 512, N_, hb, nullptr, nullptr, nullptr, nullptr, 0.0f, nullptr, 0, 0, 1, 1);
    rms_kernel<1><<<B_ * N_, 256, 0, stream>>>(hb, tb1g, tlin, nullptr, hact);
    gemm_wmma<3><<<gd, 256, 0, stream>>>(tb2h, hact, tb2b, 512, 512, 512, N_, hb, nullptr, nullptr, nullptr, nullptr, 0.0f, nullptr, 0, 0, 1, 1);
    rms_kernel<2><<<B_ * N_, 256, 0, stream>>>(hb, tb2g, nullptr, dense, (float*)d_out);
}
